// LinVarAttention_47330539602296
// MI455X (gfx1250) — matrix operand order repaired and checked by reading, compiled, not run on hardware
//
#include <hip/hip_runtime.h>
#include <hip/hip_bf16.h>

// ---------------------------------------------------------------------------
// LinVarAttention for MI455X (gfx1250): f16 WMMA (fp32 accumulate) everywhere.
// Pass 0: one-shot f32->f16 convert of x / qkv_w / proj_w (operands then load
//         as straight b128 f16 -- no per-tile re-convert, half the bytes).
// Pass 1: qkv GEMM with 2x2 register blocking (32x32 tile, 4 WMMA / K-step).
// Pass 2: single-pass unnormalized attention (t=(s*scale+5)^2 >= 0, accumulate
//         t@V and rowsum together; divide once at the end).
// Pass 3: proj GEMM with 2x2 register blocking.
// ---------------------------------------------------------------------------

typedef _Float16 half_t;
typedef __attribute__((ext_vector_type(16))) _Float16 v16h;
typedef __attribute__((ext_vector_type(8)))  _Float16 v8h;
typedef __attribute__((ext_vector_type(8)))  float    v8f;
typedef __attribute__((ext_vector_type(4)))  float    v4f;

#define BB 8
#define NN 1024
#define DD 256
#define HH 8
#define DH 32

__device__ __forceinline__ v8f wmma_f16(v16h a, v16h b, v8f c) {
  // D = A(16x32 f16) * B(32x16 f16) + C(16x16 f32)
  return __builtin_amdgcn_wmma_f32_16x16x32_f16(false, a, false, b,
                                                (short)0, c, false, false);
}

// A-operand lane layout (16-bit, 16x32):  lane = {khalf, M=lane&15};
//   elements 0..7  = K  khalf*8 .. +7,  elements 8..15 = K 16+khalf*8 .. +7
// B-operand lane layout (32x16):          lane = {khalf, N=lane&15};
//   this code loads A and B with K khalf*16 .. +15 in elements 0..15 (contiguous)
// C/D layout: VGPR r, lanes 0-15 -> (M=r, N=lane); lanes 16-31 -> (M=8+r).

// ---------------------------------------------------------------------------
// Pass 0: elementwise f32 -> f16 (n must be a multiple of 2048)
// ---------------------------------------------------------------------------
__global__ __launch_bounds__(256) void cvt_kernel(
    const float* __restrict__ in, half_t* __restrict__ out)
{
  const size_t i = ((size_t)blockIdx.x * 256 + threadIdx.x) * 8;
  v4f f0 = *(const v4f*)(in + i);
  v4f f1 = *(const v4f*)(in + i + 4);
  v8h h;
#pragma unroll
  for (int j = 0; j < 4; ++j) { h[j] = (half_t)f0[j]; h[4 + j] = (half_t)f1[j]; }
  *(v8h*)(out + i) = h;
}

// ---------------------------------------------------------------------------
// Pass 1: qkv = x @ qkv_w^T + qkv_b, 32x32 output tile per wave (2x2 WMMA).
// Column tile spans exactly one head (32 == dh); q/k/v choice and head index
// are tile-uniform -> readfirstlane + scalar branch in the epilogue.
// Writes Q[b,h,n,dh], K[b,h,n,dh], V^T[b,h,dh,n] as f16.
// ---------------------------------------------------------------------------
__global__ __launch_bounds__(256) void qkv_kernel(
    const half_t* __restrict__ x, const half_t* __restrict__ w,
    const float* __restrict__ bias,
    half_t* __restrict__ qh, half_t* __restrict__ kh, half_t* __restrict__ vt)
{
  const int lane = threadIdx.x & 31;
  const int wid  = blockIdx.x * 8 + (threadIdx.x >> 5);
  const int mt   = wid / 24;           // 256 row tiles of 32 (B*N/32)
  const int nt   = wid % 24;           // 24 col tiles of 32 (3*D/32)
  const int c16  = lane & 15;
  const int kh2  = lane >> 4;

  const half_t* arow0 = x + (size_t)(mt * 32 + c16) * DD;
  const half_t* arow1 = arow0 + (size_t)16 * DD;
  const half_t* brow0 = w + (size_t)(nt * 32 + c16) * DD;
  const half_t* brow1 = brow0 + (size_t)16 * DD;

  v8f acc00 = {}, acc01 = {}, acc10 = {}, acc11 = {};
#pragma unroll 2
  for (int k0 = 0; k0 < DD; k0 += 32) {
    v16h a0, a1;
    {
      v8h x0 = *(const v8h*)(arow0 + k0 + kh2 * 16);
      v8h x1 = *(const v8h*)(arow0 + k0 + 8 + kh2 * 16);
      v8h y0 = *(const v8h*)(arow1 + k0 + kh2 * 16);
      v8h y1 = *(const v8h*)(arow1 + k0 + 8 + kh2 * 16);
#pragma unroll
      for (int i = 0; i < 8; ++i) {
        a0[i] = x0[i]; a0[8 + i] = x1[i];
        a1[i] = y0[i]; a1[8 + i] = y1[i];
      }
    }
    v16h b0 = *(const v16h*)(brow0 + k0 + kh2 * 16);
    v16h b1 = *(const v16h*)(brow1 + k0 + kh2 * 16);
    acc00 = wmma_f16(a0, b0, acc00);
    acc01 = wmma_f16(a0, b1, acc01);
    acc10 = wmma_f16(a1, b0, acc10);
    acc11 = wmma_f16(a1, b1, acc11);
  }

  const int colbase = nt * 32;          // [0,768): one full head of q|k|v
  const int wch  = __builtin_amdgcn_readfirstlane(colbase >> 8); // 0=q 1=k 2=v
  const int hidx = (colbase & 255) >> 5;
  const float bs0 = bias[colbase + c16];
  const float bs1 = bias[colbase + 16 + c16];

  half_t h00[8], h01[8], h10[8], h11[8];
#pragma unroll
  for (int r = 0; r < 8; ++r) {
    h00[r] = (half_t)(acc00[r] + bs0);
    h01[r] = (half_t)(acc01[r] + bs1);
    h10[r] = (half_t)(acc10[r] + bs0);
    h11[r] = (half_t)(acc11[r] + bs1);
  }

  const int row0 = mt * 32 + kh2 * 8;   // rows row0..+7 (acc*0) / +16 (acc1*)
  const int b    = row0 >> 10;          // uniform within wave (32 | 1024)
  const int n0   = row0 & 1023;
  const size_t bh = (size_t)(b * HH + hidx);

  if (wch != 2) {
    half_t* base = (wch == 0 ? qh : kh) + (bh * NN + n0) * DH;
#pragma unroll
    for (int r = 0; r < 8; ++r) {
      base[(size_t)r * DH + c16]             = h00[r];
      base[(size_t)r * DH + 16 + c16]        = h01[r];
      base[(size_t)(16 + r) * DH + c16]      = h10[r];
      base[(size_t)(16 + r) * DH + 16 + c16] = h11[r];
    }
  } else {
    half_t* base = vt + bh * DH * NN + n0;
#pragma unroll
    for (int r = 0; r < 8; ++r) {
      base[(size_t)c16 * NN + r]             = h00[r];
      base[(size_t)(16 + c16) * NN + r]      = h01[r];
      base[(size_t)c16 * NN + 16 + r]        = h10[r];
      base[(size_t)(16 + c16) * NN + 16 + r] = h11[r];
    }
  }
}

// ---------------------------------------------------------------------------
// Pass 2: streaming attention. One wave = 16 queries x all 1024 keys for one
// (b,h).  Per 32-key block: 2 score WMMAs (dh=32 is exactly one K step),
// t=(s*scale+5)^2, accumulate rowsum, transpose t (C-layout -> A-layout) via a
// per-wave LDS slab guarded by s_wait_dscnt, then 2 WMMAs accumulating t @ V.
// ---------------------------------------------------------------------------
__global__ __launch_bounds__(256) void attn_kernel(
    const half_t* __restrict__ qh, const half_t* __restrict__ kh,
    const half_t* __restrict__ vt, half_t* __restrict__ o)
{
  __shared__ __align__(16) half_t tbuf[8][16 * 32];  // per-wave 1KB slab
  const int lane = threadIdx.x & 31;
  const int wv   = threadIdx.x >> 5;
  const int wid  = blockIdx.x * 8 + wv;
  const int bh   = wid >> 6;     // (b*H + h) in [0,64)
  const int qt   = wid & 63;     // query tile in [0,64)
  const int c16  = lane & 15;
  const int kh2  = lane >> 4;
  const float scale = 0.17677669529663687f;  // 1/sqrt(32)

  // Q tile A-operand, constant across the key loop
  const half_t* qrow = qh + ((size_t)bh * NN + qt * 16 + c16) * DH;
  v16h aQ;
  {
    v8h q0 = *(const v8h*)(qrow + kh2 * 16);
    v8h q1 = *(const v8h*)(qrow + 8 + kh2 * 16);
#pragma unroll
    for (int i = 0; i < 8; ++i) { aQ[i] = q0[i]; aQ[8 + i] = q1[i]; }
  }

  // loop-invariant base pointers
  const half_t* kbase  = kh + ((size_t)bh * NN + c16) * DH + kh2 * 16;
  const half_t* vbase0 = vt + ((size_t)bh * DH +      c16) * NN + kh2 * 16;
  const half_t* vbase1 = vt + ((size_t)bh * DH + 16 + c16) * NN + kh2 * 16;

  v8f oacc0 = {}, oacc1 = {}, rowacc = {};
  half_t* slab = tbuf[wv];

  for (int kb = 0; kb < NN; kb += 32) {
    if (kb + 32 < NN)  // global_prefetch_b8 on next key block
      __builtin_prefetch(kbase + (size_t)(kb + 32) * DH, 0, 3);

#pragma unroll
    for (int sub = 0; sub < 2; ++sub) {
      const half_t* krow = kbase + (size_t)(kb + sub * 16) * DH;
      v16h bK;
      v8h k0v = *(const v8h*)(krow);
      v8h k1v = *(const v8h*)(krow + 8);
#pragma unroll
      for (int i = 0; i < 8; ++i) { bK[i] = k0v[i]; bK[8 + i] = k1v[i]; }
      v8f s = {};
      s = wmma_f16(aQ, bK, s);
#pragma unroll
      for (int r = 0; r < 8; ++r) {
        float t = s[r] * scale + 5.0f;   // C_SHIFT
        t = t * t;
        rowacc[r] += t;
        slab[(kh2 * 8 + r) * 32 + sub * 16 + c16] = (half_t)t;
      }
    }
    asm volatile("s_wait_dscnt 0" ::: "memory");  // wave-local LDS transpose
    __builtin_amdgcn_wave_barrier();
    v16h aT;
    {
      const half_t* trow = slab + c16 * 32;
      v8h t0 = *(const v8h*)(trow + kh2 * 16);
      v8h t1 = *(const v8h*)(trow + 8 + kh2 * 16);
#pragma unroll
      for (int i = 0; i < 8; ++i) { aT[i] = t0[i]; aT[8 + i] = t1[i]; }
    }
    // B operands: V block (32 keys x 32 dh) from transposed [dh][key] layout
    v16h bV0 = *(const v16h*)(vbase0 + kb);
    v16h bV1 = *(const v16h*)(vbase1 + kb);
    oacc0 = wmma_f16(aT, bV0, oacc0);
    oacc1 = wmma_f16(aT, bV1, oacc1);
  }

  // rowsum: reduce across the 16-lane half holding this row's columns
#pragma unroll
  for (int r = 0; r < 8; ++r) {
    float v = rowacc[r];
    v += __shfl_xor(v, 1, 32);
    v += __shfl_xor(v, 2, 32);
    v += __shfl_xor(v, 4, 32);
    v += __shfl_xor(v, 8, 32);
    rowacc[r] = v;
  }

  const int b = bh >> 3, h = bh & 7;
  half_t* obase = o + ((size_t)b * NN + qt * 16 + kh2 * 8) * DD + h * DH;
#pragma unroll
  for (int r = 0; r < 8; ++r) {
    const float inv = 1.0f / rowacc[r];
    half_t* orow = obase + (size_t)r * DD;
    orow[c16]      = (half_t)(oacc0[r] * inv);
    orow[16 + c16] = (half_t)(oacc1[r] * inv);
  }
}

// ---------------------------------------------------------------------------
// Pass 3: out = O @ proj_w^T + proj_b, 32x32 tile per wave (2x2 WMMA).
// O and proj_w are both f16 already; output fp32.
// ---------------------------------------------------------------------------
__global__ __launch_bounds__(256) void proj_kernel(
    const half_t* __restrict__ o, const half_t* __restrict__ w,
    const float* __restrict__ bias, float* __restrict__ out)
{
  const int lane = threadIdx.x & 31;
  const int wid  = blockIdx.x * 8 + (threadIdx.x >> 5);
  const int mt   = wid >> 3;     // 256 row tiles of 32
  const int nt   = wid & 7;      // 8 col tiles of 32
  const int c16  = lane & 15;
  const int kh2  = lane >> 4;

  const half_t* arow0 = o + (size_t)(mt * 32 + c16) * DD;
  const half_t* arow1 = arow0 + (size_t)16 * DD;
  const half_t* brow0 = w + (size_t)(nt * 32 + c16) * DD;
  const half_t* brow1 = brow0 + (size_t)16 * DD;

  v8f acc00 = {}, acc01 = {}, acc10 = {}, acc11 = {};
#pragma unroll 2
  for (int k0 = 0; k0 < DD; k0 += 32) {
    v16h a0, a1;
    {
      v8h x0 = *(const v8h*)(arow0 + k0 + kh2 * 16);
      v8h x1 = *(const v8h*)(arow0 + k0 + 8 + kh2 * 16);
      v8h y0 = *(const v8h*)(arow1 + k0 + kh2 * 16);
      v8h y1 = *(const v8h*)(arow1 + k0 + 8 + kh2 * 16);
#pragma unroll
      for (int i = 0; i < 8; ++i) {
        a0[i] = x0[i]; a0[8 + i] = x1[i];
        a1[i] = y0[i]; a1[8 + i] = y1[i];
      }
    }
    v16h b0 = *(const v16h*)(brow0 + k0 + kh2 * 16);
    v16h b1 = *(const v16h*)(brow1 + k0 + kh2 * 16);
    acc00 = wmma_f16(a0, b0, acc00);
    acc01 = wmma_f16(a0, b1, acc01);
    acc10 = wmma_f16(a1, b0, acc10);
    acc11 = wmma_f16(a1, b1, acc11);
  }

  const float bs0 = bias[nt * 32 + c16];
  const float bs1 = bias[nt * 32 + 16 + c16];
  float* ob0 = out + (size_t)(mt * 32 + kh2 * 8) * DD + nt * 32 + c16;
#pragma unroll
  for (int r = 0; r < 8; ++r) {
    float* p0 = ob0 + (size_t)r * DD;
    float* p1 = ob0 + (size_t)(16 + r) * DD;
    p0[0]  = acc00[r] + bs0;
    p0[16] = acc01[r] + bs1;
    p1[0]  = acc10[r] + bs0;
    p1[16] = acc11[r] + bs1;
  }
}

// ---------------------------------------------------------------------------
extern "C" void kernel_launch(void* const* d_in, const int* in_sizes, int n_in,
                              void* d_out, int out_size, void* d_ws, size_t ws_size,
                              hipStream_t stream) {
  const float* x      = (const float*)d_in[0];
  const float* qkv_w  = (const float*)d_in[1];
  const float* qkv_b  = (const float*)d_in[2];
  const float* proj_w = (const float*)d_in[3];
  const float* proj_b = (const float*)d_in[4];
  float* out = (float*)d_out;

  // workspace (all f16, fully rewritten every call):
  //   x16   : B*N*D          = 2M
  //   wq16  : 3*D*D          = 192K
  //   wp16  : D*D            = 64K
  //   Q,K   : [B,H,N,dh]     = 2M each
  //   V^T   : [B,H,dh,N]     = 2M
  //   O     : [B,N,D]        = 2M
  const size_t elems = (size_t)BB * NN * DD;      // 2M
  half_t* x16  = (half_t*)d_ws;
  half_t* wq16 = x16  + elems;
  half_t* wp16 = wq16 + (size_t)3 * DD * DD;
  half_t* qh   = wp16 + (size_t)DD * DD;
  half_t* kh   = qh + elems;
  half_t* vt   = kh + elems;
  half_t* ob   = vt + elems;

  // Pass 0: one-shot f32->f16 of the GEMM operands (2048 elems per block)
  cvt_kernel<<<1024, 256, 0, stream>>>(x,      x16);   // 2M
  cvt_kernel<<<  96, 256, 0, stream>>>(qkv_w,  wq16);  // 192K
  cvt_kernel<<<  32, 256, 0, stream>>>(proj_w, wp16);  // 64K

  // Pass 1: 256*24 tiles / 8 waves-per-block
  qkv_kernel <<<768, 256, 0, stream>>>(x16, wq16, qkv_b, qh, kh, vt);
  // Pass 2: 64 (b,h) * 64 query tiles / 8 waves-per-block
  attn_kernel<<<512, 256, 0, stream>>>(qh, kh, vt, ob);
  // Pass 3: 256*8 tiles / 8 waves-per-block
  proj_kernel<<<256, 256, 0, stream>>>(ob, wp16, proj_b, out);
}
